// SimpleFineGrainedHead4_63883343561450
// MI455X (gfx1250) — matrix operand order repaired and checked by reading, not compiled, not run on hardware
//
#include <hip/hip_runtime.h>
#include <hip/hip_bf16.h>

// ---------- problem constants ----------
#define Bq   128   // batch (text)
#define Qq   128   // queries (visual)
#define Tt   64    // text tokens
#define Vv   196   // visual tokens
#define Dk   768   // input dim
#define Ee   512   // embed dim

// flat fp32 output offsets (return-order concatenation)
#define OFF_VCLS 0ull
#define OFF_TCLS 65536ull
#define OFF_VE   131072ull
#define OFF_TE   12976128ull
#define OFF_MASK 17170432ull
#define OFF_SIM  17178624ull

typedef __attribute__((ext_vector_type(2)))  float  v2f;
typedef __attribute__((ext_vector_type(8)))  float  v8f;
typedef __attribute__((ext_vector_type(16))) __bf16 v16bf;

union B16x16 { v16bf v; uint4 q[2]; };

__device__ __forceinline__ unsigned short f2bf(float x) {
    union { float f; unsigned u; } c; c.f = x;
    unsigned r = c.u + 0x7FFFu + ((c.u >> 16) & 1u);   // round-to-nearest-even
    return (unsigned short)(r >> 16);
}

// ======================================================================
// Tensor Data Mover: 2D tile global->LDS DMA (descriptor per ISA 08 §8)
//  - data_size = 2 bytes (bf16)
//  - LDS padding: 16B pad every 128B  -> row stride 144B (= TSTR ushorts)
//  - rows beyond tensor_dim1 are zero-filled (free V padding)
// ======================================================================
#if __has_builtin(__builtin_amdgcn_tensor_load_to_lds)
#define HAVE_TDM 1
typedef __attribute__((ext_vector_type(4))) unsigned tdm_v4u;
typedef __attribute__((ext_vector_type(8))) int      tdm_v8i;
typedef __attribute__((ext_vector_type(4))) int      tdm_v4i;

__device__ __forceinline__ void tdm_load_2d(
    unsigned lds_byte_addr, const void* gptr,
    unsigned tensor_d0, unsigned tensor_d1,
    unsigned tile_d0, unsigned tile_d1, unsigned stride0)
{
    const unsigned long long ga = (unsigned long long)gptr;
    tdm_v4u g0;
    g0[0] = 1u;                                             // count=1, user mode
    g0[1] = lds_byte_addr;                                  // lds_addr
    g0[2] = (unsigned)ga;                                   // global_addr[31:0]
    g0[3] = (unsigned)((ga >> 32) & 0x01FFFFFFu) | (2u << 30); // addr[56:32] | type=2
    tdm_v8i g1;
    //        data_size=2B   pad_en      pad_interval=32dw  pad_amount=4dw
    g1[0] = (int)((1u << 16) | (1u << 20) | (4u << 22) | (3u << 25));
    g1[1] = (int)((tensor_d0 & 0xFFFFu) << 16);             // tensor_dim0[15:0]
    g1[2] = (int)((tensor_d0 >> 16) | ((tensor_d1 & 0xFFFFu) << 16));
    g1[3] = (int)((tensor_d1 >> 16) | (tile_d0 << 16));     // tile_dim0
    g1[4] = (int)tile_d1;                                   // tile_dim1 (tile_dim2=0)
    g1[5] = (int)stride0;                                   // tensor_dim0_stride[31:0]
    g1[6] = 0; g1[7] = 0;
    tdm_v4i z4 = {0, 0, 0, 0};
#if __clang_major__ >= 23
    tdm_v8i z8 = {0, 0, 0, 0, 0, 0, 0, 0};
    __builtin_amdgcn_tensor_load_to_lds(g0, g1, z4, z4, z8, 0);
#else
    __builtin_amdgcn_tensor_load_to_lds(g0, g1, z4, z4, 0);
#endif
}
#else
#define HAVE_TDM 0
#endif

// ======================================================================
// Kernel 1: Y[M,512] = X[M,768] @ W[512,768]^T + bias   (exact fp32 WMMA)
// ======================================================================
#define XS_STR 34
__global__ __launch_bounds__(256) void gemm_bias_f32(
    const float* __restrict__ X, const float* __restrict__ W,
    const float* __restrict__ bias, float* __restrict__ Y)
{
    __shared__ __align__(16) float Xs[64 * XS_STR];
    __shared__ __align__(16) float Ws[64 * XS_STR];

    const int t    = threadIdx.x;
    const int lane = t & 31;
    const int w    = t >> 5;
    const int mt   = w & 3;
    const int nt0  = (w >> 2) * 2;
    const int l16  = lane & 15;
    const int lh   = lane >> 4;
    const size_t m0 = (size_t)blockIdx.x * 64;
    const size_t n0 = (size_t)blockIdx.y * 64;

    v8f acc0, acc1;
    {
        const float b0 = bias[n0 + nt0 * 16 + l16];
        const float b1 = bias[n0 + (nt0 + 1) * 16 + l16];
#pragma unroll
        for (int r = 0; r < 8; ++r) { acc0[r] = b0; acc1[r] = b1; }
    }

    for (int k0 = 0; k0 < Dk; k0 += 32) {
        for (int i = t; i < 512; i += 256) {
            const int row = i >> 3, c = (i & 7) * 4;
            const float4 xv = *(const float4*)(X + (m0 + row) * Dk + k0 + c);
            Xs[row * XS_STR + c + 0] = xv.x; Xs[row * XS_STR + c + 1] = xv.y;
            Xs[row * XS_STR + c + 2] = xv.z; Xs[row * XS_STR + c + 3] = xv.w;
            const float4 wv = *(const float4*)(W + (n0 + row) * Dk + k0 + c);
            Ws[row * XS_STR + c + 0] = wv.x; Ws[row * XS_STR + c + 1] = wv.y;
            Ws[row * XS_STR + c + 2] = wv.z; Ws[row * XS_STR + c + 3] = wv.w;
            if (k0 + 32 < Dk) {                           // -> global_prefetch_b8
                __builtin_prefetch(X + (m0 + row) * Dk + k0 + 32 + c, 0, 1);
                __builtin_prefetch(W + (n0 + row) * Dk + k0 + 32 + c, 0, 1);
            }
        }
        __syncthreads();
#pragma unroll
        for (int kk = 0; kk < 32; kk += 4) {
            const int kc = kk + 2 * lh;
            v2f a, ba, bb;
            a[0]  = Xs[(mt * 16 + l16) * XS_STR + kc];
            a[1]  = Xs[(mt * 16 + l16) * XS_STR + kc + 1];
            ba[0] = Ws[(nt0 * 16 + l16) * XS_STR + kc];
            ba[1] = Ws[(nt0 * 16 + l16) * XS_STR + kc + 1];
            bb[0] = Ws[((nt0 + 1) * 16 + l16) * XS_STR + kc];
            bb[1] = Ws[((nt0 + 1) * 16 + l16) * XS_STR + kc + 1];
            acc0 = __builtin_amdgcn_wmma_f32_16x16x4_f32(false, a, false, ba,
                                                         (short)0, acc0, false, false);
            acc1 = __builtin_amdgcn_wmma_f32_16x16x4_f32(false, a, false, bb,
                                                         (short)0, acc1, false, false);
        }
        __syncthreads();
    }
#pragma unroll
    for (int r = 0; r < 8; ++r) {
        const size_t m = m0 + mt * 16 + r + 8 * lh;
        Y[m * Ee + n0 + nt0 * 16 + l16]       = acc0[r];
        Y[m * Ee + n0 + (nt0 + 1) * 16 + l16] = acc1[r];
    }
}

// ======================================================================
// Kernel 2: in-place row l2-norm of Y[rows,512]; also emit bf16 copy
// ======================================================================
__global__ __launch_bounds__(128) void l2norm_bf16(
    float* __restrict__ Y, unsigned short* __restrict__ Ybf)
{
    const size_t row = blockIdx.x;
    const int t = threadIdx.x;
    float4 v = *(float4*)(Y + row * Ee + t * 4);
    float s = v.x * v.x + v.y * v.y + v.z * v.z + v.w * v.w;
#pragma unroll
    for (int m = 1; m < 32; m <<= 1) s += __shfl_xor(s, m, 32);
    __shared__ float ps[4];
    if ((t & 31) == 0) ps[t >> 5] = s;
    __syncthreads();
    s = ps[0] + ps[1] + ps[2] + ps[3];
    const float scale = 1.0f / fmaxf(sqrtf(s), 1e-12f);
    v.x *= scale; v.y *= scale; v.z *= scale; v.w *= scale;
    *(float4*)(Y + row * Ee + t * 4) = v;
    ushort4 h; h.x = f2bf(v.x); h.y = f2bf(v.y); h.z = f2bf(v.z); h.w = f2bf(v.w);
    *(ushort4*)(Ybf + row * Ee + t * 4) = h;
}

// ======================================================================
// Kernel 3: text mask [B,T]
// ======================================================================
__global__ void text_mask_k(const int* __restrict__ len, float* __restrict__ mask)
{
    const int i = blockIdx.x * blockDim.x + threadIdx.x;
    if (i < Bq * Tt) mask[i] = ((i & (Tt - 1)) < len[i >> 6]) ? 1.0f : 0.0f;
}

// ======================================================================
// Kernel 4: MaxSim. One block per (b,q); bf16 WMMA; TDM double-buffered
// staging (async global->LDS tile DMA), V padded 196->224 via TDM zero-fill.
// ======================================================================
#define KC    64
#define TSTR  72    // 64 bf16 + 8 pad = 144B row stride (TDM pad reproduces this)
__global__ __launch_bounds__(256) void maxsim(
    const unsigned short* __restrict__ te,   // [B*64, 512] bf16
    const unsigned short* __restrict__ ve,   // [Q*196, 512] bf16
    float* __restrict__ sim)                  // [B, Q]
{
    __shared__ __align__(16) unsigned short teS[2][64 * TSTR];
    __shared__ __align__(16) unsigned short veS[2][224 * TSTR];
    __shared__ float pmaxV[2][64];
    __shared__ float pmaxT[4][224];
    __shared__ float redT[64];
    __shared__ float redV[200];

    const int q = blockIdx.x, b = blockIdx.y;
    const int t = threadIdx.x, lane = t & 31, w = t >> 5;
    const int wt = w & 3, wv = w >> 2;
    const int l16 = lane & 15, lh = lane >> 4;

    v8f acc[7];
#pragma unroll
    for (int j = 0; j < 7; ++j)
#pragma unroll
        for (int r = 0; r < 8; ++r) acc[j][r] = 0.0f;

    const size_t teBase = (size_t)b * Tt * Ee;
    const size_t veBase = (size_t)q * Vv * Ee;

#if HAVE_TDM
    const unsigned teOff = (unsigned)(unsigned long long)(const void*)&teS[0][0];
    const unsigned veOff = (unsigned)(unsigned long long)(const void*)&veS[0][0];
    const unsigned teBufB = 64u * TSTR * 2u;   // bytes per te buffer
    const unsigned veBufB = 224u * TSTR * 2u;  // bytes per ve buffer
    if (t == 0) {                               // prologue: chunk 0 -> buffer 0
        tdm_load_2d(teOff, te + teBase, Ee, Tt, KC, 64, Ee);
        tdm_load_2d(veOff, ve + veBase, Ee, Vv, KC, 224, Ee);
    }
#endif

    for (int c = 0; c < Ee / KC; ++c) {
        int cur;
#if HAVE_TDM
        cur = c & 1;
        if (t == 0) {
            if (c + 1 < Ee / KC) {              // issue next chunk into other buffer
                const int nxt = cur ^ 1;
                const int k1  = (c + 1) * KC;
                tdm_load_2d(teOff + (unsigned)nxt * teBufB, te + teBase + k1,
                            Ee, Tt, KC, 64, Ee);
                tdm_load_2d(veOff + (unsigned)nxt * veBufB, ve + veBase + k1,
                            Ee, Vv, KC, 224, Ee);
                __builtin_amdgcn_s_wait_tensorcnt((short)2);  // chunk c retired
            } else {
                __builtin_amdgcn_s_wait_tensorcnt((short)0);
            }
        }
#else
        cur = 0;
        const int k0 = c * KC;
        for (int i = t; i < 512; i += 256) {              // te: 64 x 64 bf16
            const int row = i >> 3, cc = (i & 7) * 8;
            *(uint4*)&teS[0][row * TSTR + cc] =
                *(const uint4*)&te[teBase + (size_t)row * Ee + k0 + cc];
        }
        for (int i = t; i < 1792; i += 256) {             // ve: 224 x 64 bf16 (pad 0)
            const int row = i >> 3, cc = (i & 7) * 8;
            uint4 val = make_uint4(0u, 0u, 0u, 0u);
            if (row < Vv) val = *(const uint4*)&ve[veBase + (size_t)row * Ee + k0 + cc];
            *(uint4*)&veS[0][row * TSTR + cc] = val;
        }
#endif
        __syncthreads();                         // DMA/stage complete, visible to all
        const unsigned short* teB = teS[cur];
        const unsigned short* veB = veS[cur];
#pragma unroll
        for (int kk = 0; kk < KC; kk += 32) {
            B16x16 a;
            const unsigned short* ta = &teB[(wt * 16 + l16) * TSTR + kk + 8 * lh];
            a.q[0] = *(const uint4*)ta;
            a.q[1] = *(const uint4*)(ta + 16);
#pragma unroll
            for (int j = 0; j < 7; ++j) {
                B16x16 bb;
                const unsigned short* tb =
                    &veB[((wv * 7 + j) * 16 + l16) * TSTR + kk + 8 * lh];
                bb.q[0] = *(const uint4*)tb;
                bb.q[1] = *(const uint4*)(tb + 16);
                acc[j] = __builtin_amdgcn_wmma_f32_16x16x32_bf16(
                    false, a.v, false, bb.v, (short)0, acc[j], false, false);
            }
        }
        __syncthreads();                         // done reading buffer `cur`
    }

    const float NEG = -3.402823466e38f;
#pragma unroll
    for (int r = 0; r < 8; ++r) {
        float m = NEG;
#pragma unroll
        for (int j = 0; j < 7; ++j)
            if ((wv * 7 + j) * 16 + l16 < Vv) m = fmaxf(m, acc[j][r]);
        m = fmaxf(m, __shfl_xor(m, 1, 32));
        m = fmaxf(m, __shfl_xor(m, 2, 32));
        m = fmaxf(m, __shfl_xor(m, 4, 32));
        m = fmaxf(m, __shfl_xor(m, 8, 32));
        if (l16 == 0) pmaxV[wv][wt * 16 + r + 8 * lh] = m;
    }
#pragma unroll
    for (int j = 0; j < 7; ++j) {
        float m = NEG;
#pragma unroll
        for (int r = 0; r < 8; ++r) m = fmaxf(m, acc[j][r]);
        m = fmaxf(m, __shfl_xor(m, 16, 32));
        if (lane < 16) pmaxT[wt][(wv * 7 + j) * 16 + l16] = m;
    }
    __syncthreads();
    if (t < 64)  redT[t] = fmaxf(pmaxV[0][t], pmaxV[1][t]);
    if (t < Vv)  redV[t] = fmaxf(fmaxf(pmaxT[0][t], pmaxT[1][t]),
                                 fmaxf(pmaxT[2][t], pmaxT[3][t]));
    __syncthreads();
    if (t == 0) {
        float sT = 0.0f, sV = 0.0f;
        for (int i = 0; i < Tt; ++i) sT += redT[i];
        for (int i = 0; i < Vv; ++i) sV += redV[i];
        sim[(size_t)b * Qq + q] = 0.5f * (sT / (float)Vv + sV / (float)Tt);
    }
}

// ======================================================================
extern "C" void kernel_launch(void* const* d_in, const int* in_sizes, int n_in,
                              void* d_out, int out_size, void* d_ws, size_t ws_size,
                              hipStream_t stream)
{
    const float* visual_cls     = (const float*)d_in[0];
    const float* textual_cls    = (const float*)d_in[1];
    const float* visual_tokens  = (const float*)d_in[2];
    const float* textual_tokens = (const float*)d_in[3];
    const float* Wv_cls = (const float*)d_in[4];  const float* bv_cls = (const float*)d_in[5];
    const float* Wt_cls = (const float*)d_in[6];  const float* bt_cls = (const float*)d_in[7];
    const float* Wv_tok = (const float*)d_in[8];  const float* bv_tok = (const float*)d_in[9];
    const float* Wt_tok = (const float*)d_in[10]; const float* bt_tok = (const float*)d_in[11];
    const int*   text_length = (const int*)d_in[12];

    float* out = (float*)d_out;
    unsigned short* veb = (unsigned short*)d_ws;                 // [Q*196, 512] bf16
    unsigned short* teb = veb + (size_t)Qq * Vv * Ee;            // [B*64, 512]  bf16

    gemm_bias_f32<<<dim3(2, 8),   256, 0, stream>>>(visual_cls,  Wv_cls, bv_cls, out + OFF_VCLS);
    gemm_bias_f32<<<dim3(2, 8),   256, 0, stream>>>(textual_cls, Wt_cls, bt_cls, out + OFF_TCLS);
    gemm_bias_f32<<<dim3(392, 8), 256, 0, stream>>>(visual_tokens,  Wv_tok, bv_tok, out + OFF_VE);
    gemm_bias_f32<<<dim3(128, 8), 256, 0, stream>>>(textual_tokens, Wt_tok, bt_tok, out + OFF_TE);
    l2norm_bf16<<<Qq * Vv, 128, 0, stream>>>(out + OFF_VE, veb);
    l2norm_bf16<<<Bq * Tt, 128, 0, stream>>>(out + OFF_TE, teb);
    text_mask_k<<<(Bq * Tt + 255) / 256, 256, 0, stream>>>(text_length, out + OFF_MASK);
    maxsim<<<dim3(Qq, Bq), 256, 0, stream>>>(teb, veb, out + OFF_SIM);
}
